// BailingMoE_43293270343964
// MI455X (gfx1250) — compile-verified, matrix operand order checked by reading, not run on hardware
//
#include <hip/hip_runtime.h>

#define T_DIM 1024
#define H_DIM 2048
#define E_NUM 32
#define K_TOP 4
#define I_DIM 1408
#define IS_DIM 1408

typedef __attribute__((ext_vector_type(16))) __bf16 v16bf;
typedef __attribute__((ext_vector_type(8)))  float  v8f;

__device__ __forceinline__ float warp_max(float v) {
#pragma unroll
  for (int o = 16; o > 0; o >>= 1) v = fmaxf(v, __shfl_xor(v, o, 32));
  return v;
}
__device__ __forceinline__ float warp_sum(float v) {
#pragma unroll
  for (int o = 16; o > 0; o >>= 1) v += __shfl_xor(v, o, 32);
  return v;
}
// silu via hardware v_rcp_f32 (avoids IEEE div_scale/fixup slow path)
__device__ __forceinline__ float silu(float x) {
  return x * __builtin_amdgcn_rcpf(1.0f + __expf(-x));
}

// D = A*B + C, bf16 inputs, f32 accumulate (v_wmma_f32_16x16x32_bf16)
__device__ __forceinline__ v8f wmma_bf16(v16bf a, v16bf b, v8f c) {
  return __builtin_amdgcn_wmma_f32_16x16x32_bf16(false, a, false, b, (short)0, c,
                                                 false, false);
}

// Build a 16x32 (MxK) bf16 fragment from fp32 row-major memory.
// Wave32 layout: lane m (0-15) holds row m, K chunks {0..7, 16..23};
// lane m+16 holds row m, K chunks {8..15, 24..31}.
__device__ __forceinline__ v16bf load_frag_f32(const float* __restrict__ p0,
                                               int ld, int lane) {
  int r = lane & 15;
  int ch = (lane >> 4) << 3;
  const float* p = p0 + (size_t)r * ld + ch;
  v16bf f;
#pragma unroll
  for (int j = 0; j < 8; j++) {
    f[j]     = (__bf16)p[j];
    f[j + 8] = (__bf16)p[j + 16];
  }
  return f;
}

// Same fragment layout, but source is bf16 in LDS.
__device__ __forceinline__ v16bf load_frag_lds(const __bf16* __restrict__ p0,
                                               int ld, int lane) {
  int r = lane & 15;
  int ch = (lane >> 4) << 3;
  const __bf16* p = p0 + r * ld + ch;
  v16bf f;
#pragma unroll
  for (int j = 0; j < 8; j++) {
    f[j]     = p[j];
    f[j + 8] = p[j + 16];
  }
  return f;
}

// ---------------- Router: one wave per token, lane = expert ----------------
__global__ __launch_bounds__(256) void router_kernel(
    const float* __restrict__ x, const float* __restrict__ gate_w,
    int* __restrict__ topi, float* __restrict__ topw) {
  int lane = threadIdx.x & 31;
  int wid = threadIdx.x >> 5;
  int t = blockIdx.x * 8 + wid;
  if (t >= T_DIM) return;

  const float4* xr = (const float4*)(x + (size_t)t * H_DIM);
  const float4* gr = (const float4*)(gate_w + (size_t)lane * H_DIM);
  float acc = 0.f;
  for (int h = 0; h < H_DIM / 4; ++h) {
    float4 a = xr[h];
    float4 b = gr[h];
    acc += a.x * b.x + a.y * b.y + a.z * b.z + a.w * b.w;
  }
  float m = warp_max(acc);
  float p = __expf(acc - m);
  float s = warp_sum(p);
  p *= __builtin_amdgcn_rcpf(s);

  float cur = p;
  float ssum = 0.f;
  int myrank = -1;
#pragma unroll
  for (int k = 0; k < K_TOP; k++) {
    float bv = cur;
    int bi = lane;
#pragma unroll
    for (int o = 16; o > 0; o >>= 1) {
      float ov = __shfl_xor(bv, o, 32);
      int oi = __shfl_xor(bi, o, 32);
      if (ov > bv || (ov == bv && oi < bi)) { bv = ov; bi = oi; }
    }
    ssum += bv;
    if (lane == bi) { myrank = k; cur = -1.0f; }
  }
  if (myrank >= 0) {
    topi[t * K_TOP + myrank] = lane;
    topw[t * K_TOP + myrank] = p * __builtin_amdgcn_rcpf(ssum);
  }
}

// ------- Count / exclusive-scan / scatter: single wave, deterministic ------
__global__ void route_scan_kernel(const int* __restrict__ topi,
                                  const float* __restrict__ topw,
                                  int* __restrict__ counts,
                                  int* __restrict__ offsets,
                                  int* __restrict__ etok,
                                  float* __restrict__ ew) {
  int lane = threadIdx.x;  // lane == expert id (E_NUM == 32)
  int cnt = 0;
  for (int i = 0; i < T_DIM * K_TOP; i++)
    if (topi[i] == lane) cnt++;
  counts[lane] = cnt;

  int off = cnt;
#pragma unroll
  for (int o = 1; o < 32; o <<= 1) {
    int v = __shfl_up(off, o, 32);
    if (lane >= o) off += v;
  }
  off -= cnt;  // exclusive prefix
  offsets[lane] = off;

  int cur = off;
  for (int i = 0; i < T_DIM * K_TOP; i++) {
    if (topi[i] == lane) {
      etok[cur] = i >> 2;  // token id
      ew[cur] = topw[i];
      cur++;
    }
  }
}

// ------------- Shared expert up-proj: s = silu(g)*u, bf16 out --------------
__global__ __launch_bounds__(256) void shared_up_kernel(
    const float* __restrict__ x, const float* __restrict__ ws_gu,
    __bf16* __restrict__ s_buf) {
  __shared__ __bf16 xt[16 * H_DIM];  // 64 KB
  int tid = threadIdx.x;
  int t0 = blockIdx.x * 16;
  for (int idx = tid; idx < 16 * H_DIM; idx += 256) {
    int r = idx >> 11;
    int k = idx & (H_DIM - 1);
    xt[idx] = (__bf16)x[(size_t)(t0 + r) * H_DIM + k];
  }
  __syncthreads();

  int lane = tid & 31, wid = tid >> 5;
  int n = lane & 15;
  int mbase = (lane >> 4) << 3;
  for (int c = wid; c < IS_DIM / 16; c += 8) {
    v8f accg = {};
    v8f accu = {};
    const float* wg = ws_gu + (size_t)(c * 16) * H_DIM;
    const float* wu = ws_gu + (size_t)(IS_DIM + c * 16) * H_DIM;
    for (int k0 = 0; k0 < H_DIM; k0 += 32) {
      v16bf af = load_frag_lds(xt + k0, H_DIM, lane);
      v16bf bg = load_frag_f32(wg + k0, H_DIM, lane);
      v16bf bu = load_frag_f32(wu + k0, H_DIM, lane);
      accg = wmma_bf16(af, bg, accg);
      accu = wmma_bf16(af, bu, accu);
    }
#pragma unroll
    for (int j = 0; j < 8; j++) {
      int m = mbase + j;
      float v = silu(accg[j]) * accu[j];
      s_buf[(size_t)(t0 + m) * IS_DIM + c * 16 + n] = (__bf16)v;
    }
  }
}

// --------- Shared expert down-proj: writes (initializes) full d_out --------
__global__ __launch_bounds__(256) void shared_down_kernel(
    const __bf16* __restrict__ s_buf, const float* __restrict__ ws_d,
    float* __restrict__ out) {
  __shared__ __bf16 st[16 * IS_DIM];  // 44 KB
  int tid = threadIdx.x;
  int t0 = blockIdx.x * 16;
  for (int idx = tid; idx < 16 * IS_DIM; idx += 256) {
    int r = idx / IS_DIM;
    int k = idx - r * IS_DIM;
    st[idx] = s_buf[(size_t)(t0 + r) * IS_DIM + k];
  }
  __syncthreads();

  int lane = tid & 31, wid = tid >> 5;
  int n = lane & 15;
  int mbase = (lane >> 4) << 3;
  for (int c = wid; c < H_DIM / 16; c += 8) {
    v8f acc = {};
    const float* wd = ws_d + (size_t)(c * 16) * IS_DIM;
    for (int k0 = 0; k0 < IS_DIM; k0 += 32) {
      v16bf af = load_frag_lds(st + k0, IS_DIM, lane);
      v16bf bf = load_frag_f32(wd + k0, IS_DIM, lane);
      acc = wmma_bf16(af, bf, acc);
    }
#pragma unroll
    for (int j = 0; j < 8; j++) {
      int m = mbase + j;
      out[(size_t)(t0 + m) * H_DIM + c * 16 + n] = acc[j];
    }
  }
}

// -------- Expert up-proj: gathered tokens, a = silu(h1)*h3 * weight --------
__global__ __launch_bounds__(256) void expert_up_kernel(
    const float* __restrict__ x, const float* __restrict__ w1,
    const float* __restrict__ w3, const int* __restrict__ counts,
    const int* __restrict__ offsets, const int* __restrict__ etok,
    const float* __restrict__ ew, __bf16* __restrict__ a_buf) {
  __shared__ __bf16 xt[16 * H_DIM];  // 64 KB
  int e = blockIdx.x;
  int rt = blockIdx.y;
  int cnt = counts[e] - rt * 16;
  if (cnt <= 0) return;
  if (cnt > 16) cnt = 16;
  int base = offsets[e] + rt * 16;
  int tid = threadIdx.x;

  for (int idx = tid; idx < 16 * H_DIM; idx += 256) {
    int r = idx >> 11;
    int k = idx & (H_DIM - 1);
    int tok = (r < cnt) ? etok[base + r] : 0;
    xt[idx] = (__bf16)x[(size_t)tok * H_DIM + k];
  }
  __syncthreads();

  int lane = tid & 31, wid = tid >> 5;
  int n = lane & 15;
  int mbase = (lane >> 4) << 3;
  const float* w1e = w1 + (size_t)e * I_DIM * H_DIM;
  const float* w3e = w3 + (size_t)e * I_DIM * H_DIM;
  for (int c = wid; c < I_DIM / 16; c += 8) {
    v8f a1 = {};
    v8f a3 = {};
    const float* p1 = w1e + (size_t)(c * 16) * H_DIM;
    const float* p3 = w3e + (size_t)(c * 16) * H_DIM;
    for (int k0 = 0; k0 < H_DIM; k0 += 32) {
      v16bf af = load_frag_lds(xt + k0, H_DIM, lane);
      v16bf b1 = load_frag_f32(p1 + k0, H_DIM, lane);
      v16bf b3 = load_frag_f32(p3 + k0, H_DIM, lane);
      a1 = wmma_bf16(af, b1, a1);
      a3 = wmma_bf16(af, b3, a3);
    }
#pragma unroll
    for (int j = 0; j < 8; j++) {
      int m = mbase + j;
      if (m < cnt) {
        float w = ew[base + m];
        float v = silu(a1[j]) * a3[j] * w;
        a_buf[(size_t)(base + m) * I_DIM + c * 16 + n] = (__bf16)v;
      }
    }
  }
}

// ------- Expert down-proj: out[tok] += a @ w2[e].T (weight pre-applied) -----
__global__ __launch_bounds__(256) void expert_down_kernel(
    const __bf16* __restrict__ a_buf, const float* __restrict__ w2,
    const int* __restrict__ counts, const int* __restrict__ offsets,
    const int* __restrict__ etok, float* __restrict__ out) {
  __shared__ __bf16 at[16 * I_DIM];  // 44 KB
  __shared__ int stok[16];
  int e = blockIdx.x;
  int rt = blockIdx.y;
  int cnt = counts[e] - rt * 16;
  if (cnt <= 0) return;
  if (cnt > 16) cnt = 16;
  int base = offsets[e] + rt * 16;
  int tid = threadIdx.x;

  if (tid < 16) stok[tid] = (tid < cnt) ? etok[base + tid] : 0;
  for (int idx = tid; idx < 16 * I_DIM; idx += 256) {
    int r = idx / I_DIM;
    int k = idx - r * I_DIM;
    at[idx] = (r < cnt) ? a_buf[(size_t)(base + r) * I_DIM + k] : (__bf16)0.f;
  }
  __syncthreads();

  int lane = tid & 31, wid = tid >> 5;
  int n = lane & 15;
  int mbase = (lane >> 4) << 3;
  const float* w2e = w2 + (size_t)e * H_DIM * I_DIM;
  for (int c = wid; c < H_DIM / 16; c += 8) {
    v8f acc = {};
    const float* pw = w2e + (size_t)(c * 16) * I_DIM;
    for (int k0 = 0; k0 < I_DIM; k0 += 32) {
      v16bf af = load_frag_lds(at + k0, I_DIM, lane);
      v16bf bf = load_frag_f32(pw + k0, I_DIM, lane);
      acc = wmma_bf16(af, bf, acc);
    }
#pragma unroll
    for (int j = 0; j < 8; j++) {
      int m = mbase + j;
      if (m < cnt) {
        int tok = stok[m];
        atomicAdd(&out[(size_t)tok * H_DIM + c * 16 + n], acc[j]);
      }
    }
  }
}

extern "C" void kernel_launch(void* const* d_in, const int* in_sizes, int n_in,
                              void* d_out, int out_size, void* d_ws,
                              size_t ws_size, hipStream_t stream) {
  (void)in_sizes; (void)n_in; (void)out_size; (void)ws_size;
  const float* x      = (const float*)d_in[0];
  const float* gate_w = (const float*)d_in[1];
  const float* w1     = (const float*)d_in[2];
  const float* w3     = (const float*)d_in[3];
  const float* w2     = (const float*)d_in[4];
  const float* ws_gu  = (const float*)d_in[5];
  const float* ws_d   = (const float*)d_in[6];
  float* out = (float*)d_out;

  // Workspace layout (~15.5 MB total)
  char* ws = (char*)d_ws;
  int*    topi    = (int*)(ws);                    // 16 KB
  float*  topw    = (float*)(ws + (16 << 10));     // 16 KB
  int*    etok    = (int*)(ws + (32 << 10));       // 16 KB
  float*  ew      = (float*)(ws + (48 << 10));     // 16 KB
  int*    counts  = (int*)(ws + (64 << 10));       // 128 B
  int*    offsets = (int*)(ws + (64 << 10) + 256); // 128 B
  __bf16* s_buf   = (__bf16*)(ws + (128 << 10));   // T*IS bf16 = 2.75 MB
  __bf16* a_buf   = (__bf16*)(ws + (4 << 20));     // T*K*I bf16 = 11.5 MB

  router_kernel<<<T_DIM / 8, 256, 0, stream>>>(x, gate_w, topi, topw);
  route_scan_kernel<<<1, 32, 0, stream>>>(topi, topw, counts, offsets, etok, ew);
  shared_up_kernel<<<T_DIM / 16, 256, 0, stream>>>(x, ws_gu, s_buf);
  shared_down_kernel<<<T_DIM / 16, 256, 0, stream>>>(s_buf, ws_d, out);
  expert_up_kernel<<<dim3(E_NUM, T_DIM / 16), 256, 0, stream>>>(
      x, w1, w3, counts, offsets, etok, ew, a_buf);
  expert_down_kernel<<<dim3(E_NUM, T_DIM / 16), 256, 0, stream>>>(
      a_buf, w2, counts, offsets, etok, out);
}
